// HdcLevelEncoder_3874060501325
// MI455X (gfx1250) — compile-verified, matrix operand order checked by reading, not run on hardware
//
#include <hip/hip_runtime.h>
#include <hip/hip_bf16.h>

// ---------------------------------------------------------------------------
// HDC level encoder for MI455X (gfx1250, wave32).
// Dominant work: column-sum over T=2048 of elementwise ±{1,3} products for
// DIM=10000 columns.  ~200MB unique HBM traffic => ~9us roofline; we stream
// through L2 and use V_WMMA_F32_16X16X32_F16 (A = ones) as the K-reduction
// engine.  f16 products are exact here ({-3,-1,1,3}), accumulate in f32.
// ---------------------------------------------------------------------------

#define TSTAMPS 2048
#define DIM     10000
#define LEVELS  1024
#define NCHUNK  8          // T split into 8 chunks of 256
#define CHUNK_T 256

typedef __attribute__((ext_vector_type(16))) _Float16 v16h;
typedef __attribute__((ext_vector_type(8)))  float    v8f;

// --------------------------- K1: index precompute ---------------------------
__global__ void hdc_indices(const float* __restrict__ input,
                            int* __restrict__ tidx,
                            int* __restrict__ eidx)
{
    int t = blockIdx.x * blockDim.x + threadIdx.x;
    if (t >= TSTAMPS) return;

    // value_to_index(time, 0, 2048, 2048): round(x/2048*2047), clip [0,2047]
    float tv = input[t * 4 + 0];
    float fi = rintf(tv * (2047.0f / 2048.0f));     // RNE matches jnp.round
    fi = fminf(fmaxf(fi, 0.0f), 2047.0f);
    tidx[t] = (int)fi;

#pragma unroll
    for (int c = 0; c < 3; ++c) {
        // clip to [-5,5]; value_to_index(x,-5,5,1024): round((x+5)/10*1023)
        float x = input[t * 4 + 1 + c];
        x = fminf(fmaxf(x, -5.0f), 5.0f);
        float e = rintf((x + 5.0f) * (1023.0f / 10.0f));
        e = fminf(fmaxf(e, 0.0f), 1023.0f);
        eidx[3 * t + c] = (int)e;
    }
}

// --------------------------- K2: sinusoid factor ----------------------------
__device__ __forceinline__ float sinu(const float* __restrict__ w,
                                      const float* __restrict__ b,
                                      const float* __restrict__ x,
                                      int k, int d)
{
    float p = 0.0f;
#pragma unroll 6
    for (int j = 0; j < k; ++j) p = fmaf(w[d * k + j], x[j], p);
    return cosf(p + b[d]) * sinf(p);
}

__global__ void hdc_factor(const float* __restrict__ feat,
                           const float* __restrict__ w_rms,  const float* __restrict__ b_rms,
                           const float* __restrict__ w_mfcc, const float* __restrict__ b_mfcc,
                           const float* __restrict__ w_fm,   const float* __restrict__ b_fm,
                           const float* __restrict__ w_fx,   const float* __restrict__ b_fx,
                           const float* __restrict__ w_fv,   const float* __restrict__ b_fv,
                           const float* __restrict__ w_mean, const float* __restrict__ b_mean,
                           const float* __restrict__ w_max,  const float* __restrict__ b_max,
                           const float* __restrict__ w_var,  const float* __restrict__ b_var,
                           const float* __restrict__ w_sc,   const float* __restrict__ b_sc,
                           float* __restrict__ factor)
{
    int d = blockIdx.x * blockDim.x + threadIdx.x;
    if (d >= DIM) return;

    float f_rms  = sinu(w_rms,  b_rms,  feat + 0,  3, d);
    float f_mfcc = sinu(w_mfcc, b_mfcc, feat + 3,  6, d);
    float f_fm   = sinu(w_fm,   b_fm,   feat + 9,  3, d);
    float f_fx   = sinu(w_fx,   b_fx,   feat + 12, 3, d);
    float f_fv   = sinu(w_fv,   b_fv,   feat + 15, 3, d);
    float f_mean = sinu(w_mean, b_mean, feat + 18, 3, d);
    float f_max  = sinu(w_max,  b_max,  feat + 21, 3, d);
    float f_var  = sinu(w_var,  b_var,  feat + 24, 3, d);
    float f_sc   = sinu(w_sc,   b_sc,   feat + 27, 3, d);

    factor[d] = f_rms * (f_mfcc + f_sc) * (f_fm + f_fx + f_fv)
                      * (f_mean + f_max + f_var);
}

// ------------------- K3: WMMA column-sum over timestamps --------------------
// grid = (625 d-tiles, 8 T-chunks), block = 32 (one wave).
// B tile (32x16, K=t, N=d), k order used here (A = ones, any order works): lanes 0-15 hold K=0..15,
// lanes 16-31 hold K=16..31; each lane packs its 16 K-values into v16h.
// A = all ones -> every row of D is the column sum.
__global__ __launch_bounds__(32)
void hdc_reduce_wmma(const float* __restrict__ keys,
                     const float* __restrict__ embed,
                     const float* __restrict__ ts,
                     const int*   __restrict__ tidx,
                     const int*   __restrict__ eidx,
                     float*       __restrict__ partial)
{
    const int lane  = threadIdx.x & 31;
    const int n     = lane & 15;               // d within 16-wide tile
    const int kBase = (lane >> 4) << 4;        // 0 or 16: K-half of B tile
    const int d0    = blockIdx.x * 16;
    const int d     = d0 + n;                  // DIM % 16 == 0, always in range
    const int tBase = blockIdx.y * CHUNK_T;

    v16h aOnes;
#pragma unroll
    for (int j = 0; j < 16; ++j) aOnes[j] = (_Float16)1.0f;

    v8f acc = {};

    for (int t0 = tBase; t0 < tBase + CHUNK_T; t0 += 32) {
        v16h b;
#pragma unroll
        for (int j = 0; j < 16; ++j) {
            const int t  = t0 + kBase + j;
            const int i0 = eidx[3 * t + 0];
            const int i1 = eidx[3 * t + 1];
            const int i2 = eidx[3 * t + 2];
            const int tr = tidx[t];
            float lvl = embed[i0 * DIM + d]
                      + embed[i1 * DIM + d]
                      + embed[i2 * DIM + d];            // odd integer in {-3..3}
            float prod = lvl * keys[t * DIM + d] * ts[tr * DIM + d];
            b[j] = (_Float16)prod;                      // exact in f16
        }
        // D = ones(16x32) * B(32x16) + C : every row of D = column sums
        acc = __builtin_amdgcn_wmma_f32_16x16x32_f16(
                  /*neg_a=*/false, aOnes, /*neg_b=*/false, b,
                  /*c_mod=*/(short)0, acc, /*reuse_a=*/false, /*reuse_b=*/false);
    }

    // Row 0 of D lives in acc[0]: lanes 0..15 hold columns n=0..15.
    if (lane < 16)
        partial[blockIdx.y * DIM + d0 + lane] = acc[0];
}

// ----------------------------- K4: finalize ---------------------------------
__global__ void hdc_finalize(const float* __restrict__ partial,
                             const float* __restrict__ factor,
                             float* __restrict__ out)
{
    int d = blockIdx.x * blockDim.x + threadIdx.x;
    if (d >= DIM) return;
    float s = 0.0f;
#pragma unroll
    for (int c = 0; c < NCHUNK; ++c) s += partial[c * DIM + d];   // exact ints
    float v = s * factor[d];
    out[d] = (v > 0.0f) ? 1.0f : -1.0f;
}

// ----------------------------- launch ---------------------------------------
extern "C" void kernel_launch(void* const* d_in, const int* in_sizes, int n_in,
                              void* d_out, int out_size, void* d_ws, size_t ws_size,
                              hipStream_t stream)
{
    const float* input = (const float*)d_in[0];   // (2048, 4)
    const float* feat  = (const float*)d_in[1];   // (30,)
    const float* keys  = (const float*)d_in[2];   // (2048, 10000)
    const float* embed = (const float*)d_in[3];   // (1024, 10000)
    const float* ts    = (const float*)d_in[4];   // (2048, 10000)
    const float* w_rms  = (const float*)d_in[5];  const float* b_rms  = (const float*)d_in[6];
    const float* w_mfcc = (const float*)d_in[7];  const float* b_mfcc = (const float*)d_in[8];
    const float* w_fm   = (const float*)d_in[9];  const float* b_fm   = (const float*)d_in[10];
    const float* w_fx   = (const float*)d_in[11]; const float* b_fx   = (const float*)d_in[12];
    const float* w_fv   = (const float*)d_in[13]; const float* b_fv   = (const float*)d_in[14];
    const float* w_mean = (const float*)d_in[15]; const float* b_mean = (const float*)d_in[16];
    const float* w_max  = (const float*)d_in[17]; const float* b_max  = (const float*)d_in[18];
    const float* w_var  = (const float*)d_in[19]; const float* b_var  = (const float*)d_in[20];
    const float* w_sc   = (const float*)d_in[21]; const float* b_sc   = (const float*)d_in[22];
    float* out = (float*)d_out;

    // Workspace layout (all overwritten every call; no pre-zero needed):
    //   tidx:    int[2048]
    //   eidx:    int[2048*3]
    //   factor:  float[10000]
    //   partial: float[8*10000]
    int*   tidx    = (int*)d_ws;
    int*   eidx    = tidx + TSTAMPS;
    float* factor  = (float*)(eidx + TSTAMPS * 3);
    float* partial = factor + ((DIM + 63) & ~63);

    hdc_indices<<<TSTAMPS / 256, 256, 0, stream>>>(input, tidx, eidx);

    hdc_factor<<<(DIM + 255) / 256, 256, 0, stream>>>(
        feat, w_rms, b_rms, w_mfcc, b_mfcc, w_fm, b_fm, w_fx, b_fx,
        w_fv, b_fv, w_mean, b_mean, w_max, b_max, w_var, b_var,
        w_sc, b_sc, factor);

    dim3 grid(DIM / 16, NCHUNK);   // 625 x 8 single-wave blocks
    hdc_reduce_wmma<<<grid, 32, 0, stream>>>(keys, embed, ts, tidx, eidx, partial);

    hdc_finalize<<<(DIM + 255) / 256, 256, 0, stream>>>(partial, factor, out);
}
